// GroupQueryAttention_23948737642845
// MI455X (gfx1250) — matrix operand order repaired and checked by reading, not compiled, not run on hardware
//
#include <hip/hip_runtime.h>
#include <hip/hip_bf16.h>

// ---------------------------------------------------------------------------
// GQA attention for gfx1250 (MI455X): all GEMMs on v_wmma_f32_16x16x32_f16,
// global->LDS staging via GLOBAL_LOAD_ASYNC_TO_LDS_B128 (ASYNCcnt).
// Shapes fixed by the reference: B=2, T=2048, E=1024, H=16, KV=4, D=64, G=4.
// ---------------------------------------------------------------------------

#define B_  2
#define T_  2048
#define E_  1024
#define H_  16
#define KV_ 4
#define D_  64
#define KVE_ (KV_ * D_)   // 256
#define BT_  (B_ * T_)    // 4096

#define USE_ASYNC_LDS 1   // set 0 to fall back to plain uint4 staging

typedef _Float16 v16h __attribute__((ext_vector_type(16)));
typedef float    v8f  __attribute__((ext_vector_type(8)));
typedef unsigned int v8u __attribute__((ext_vector_type(8)));

union FragU { v16h h; v8u u; };

// Low 32 bits of a generic pointer to __shared__ == LDS byte offset
// (AS3->AS0 cast is {aperture_hi, lds_offset} per ISA 10.2 aperture table).
__device__ __forceinline__ unsigned lds_off(const void* p) {
  return (unsigned)(unsigned long long)p;
}

// Async global->LDS 16B copy, tracked by ASYNCcnt (ISA 10.x / 08_async_tensor).
__device__ __forceinline__ void async_copy16(void* ldsDst, const void* gSrc) {
#if USE_ASYNC_LDS
  asm volatile("global_load_async_to_lds_b128 %0, %1, off"
               :: "v"(lds_off(ldsDst)), "v"(gSrc) : "memory");
#else
  *(uint4*)ldsDst = *(const uint4*)gSrc;
#endif
}

__device__ __forceinline__ void async_wait0() {
#if USE_ASYNC_LDS
  asm volatile("s_wait_asynccnt 0x0" ::: "memory");
#endif
}

// A-fragment (16x32 f16, M x K): lane m = lane&15 fixed row; hh = lane>>4.
// VGPR v: K = 16*hh + 2v.
__device__ __forceinline__ v16h load_afrag(const _Float16* rowPtr, int hh) {
  FragU f;
#pragma unroll
  for (int v = 0; v < 8; ++v)
    f.u[v] = *(const unsigned int*)(rowPtr + 16 * hh + 2 * v);
  return f.h;
}

// B-fragment (32x16 f16, K x N) gathered from a [n][k] row-major tile:
// lane n = lane&15 fixed row; VGPR v: K = 16*hh + 2v.
__device__ __forceinline__ v16h load_bfrag(const _Float16* rowPtr, int hh) {
  FragU f;
#pragma unroll
  for (int v = 0; v < 8; ++v)
    f.u[v] = *(const unsigned int*)(rowPtr + 16 * hh + 2 * v);
  return f.h;
}

__device__ __forceinline__ v8f wmma32(v16h a, v16h b, v8f c) {
  return __builtin_amdgcn_wmma_f32_16x16x32_f16(false, a, false, b,
                                                (short)0, c, false, false);
}

// ---------------------------------------------------------------------------
// Weight prep: W f32 [K][N] (row-major) -> WT f16 [N][K] (row-major)
// ---------------------------------------------------------------------------
__global__ void transpose_w(const float* __restrict__ W, _Float16* __restrict__ WT,
                            int K, int N) {
  int idx = blockIdx.x * 256 + threadIdx.x;
  if (idx >= K * N) return;
  int n = idx % N, k = idx / N;
  WT[(size_t)n * K + k] = (_Float16)W[(size_t)k * N + n];
}

// ---------------------------------------------------------------------------
// Tiled GEMM: out[M][N] = A[M][Kd](f32) @ BT[N][Kd]^T(f16) + bias, optional
// fused RoPE (per 64-wide head), f16 or f32 store.
// Block = 128x64 (4 waves, each 32 rows x 64 cols -> B-frags reused 2x).
// ---------------------------------------------------------------------------
template <bool ROPE, bool F16OUT>
__global__ __launch_bounds__(128) void gqa_gemm(
    const float* __restrict__ A, const _Float16* __restrict__ BT,
    const float* __restrict__ bias, void* __restrict__ outP,
    int M, int N, int Kd, int Tdim,
    const float* __restrict__ cosE, const float* __restrict__ sinE) {
  __shared__ alignas(16) _Float16 As[128][32];
  __shared__ alignas(16) _Float16 Bs[64][32];

  const int tid  = threadIdx.x;
  const int wave = tid >> 5;
  const int lane = tid & 31;
  const int m16  = lane & 15;
  const int hh   = lane >> 4;
  const int mBase = blockIdx.y * 128;
  const int nBase = blockIdx.x * 64;

  v8f acc[2][4];
#pragma unroll
  for (int rt = 0; rt < 2; ++rt)
#pragma unroll
    for (int nt = 0; nt < 4; ++nt)
#pragma unroll
      for (int r = 0; r < 8; ++r) acc[rt][nt][r] = 0.0f;

  for (int k0 = 0; k0 < Kd; k0 += 32) {
    __syncthreads();
    // Stage B tile (f16 [n][k]) via async global->LDS: 64x32 halfs = 256x16B.
#pragma unroll
    for (int i = 0; i < 2; ++i) {
      int u = tid + i * 128;
      int n = u >> 2, seg = u & 3;
      async_copy16(&Bs[n][seg * 8],
                   BT + (size_t)(nBase + n) * Kd + k0 + seg * 8);
    }
    // Stage A tile (f32 -> f16 convert): 128 rows x 32 cols = 1024 float4.
#pragma unroll
    for (int i = 0; i < 8; ++i) {
      int u = tid + i * 128;
      int m = u >> 3, seg = u & 7;
      float4 f = *((const float4*)(A + (size_t)(mBase + m) * Kd + k0) + seg);
      _Float16* dst = &As[m][seg * 4];
      dst[0] = (_Float16)f.x; dst[1] = (_Float16)f.y;
      dst[2] = (_Float16)f.z; dst[3] = (_Float16)f.w;
    }
    if (k0 + 32 < Kd) {  // pull next A K-tile toward L2 (global_prefetch_b8)
      __builtin_prefetch(A + (size_t)(mBase + tid) * Kd + k0 + 32, 0, 1);
    }
    async_wait0();
    __syncthreads();

    v16h af0 = load_afrag(&As[wave * 32 + m16][0], hh);
    v16h af1 = load_afrag(&As[wave * 32 + 16 + m16][0], hh);
#pragma unroll
    for (int nt = 0; nt < 4; ++nt) {
      v16h bf = load_bfrag(&Bs[nt * 16 + m16][0], hh);
      acc[0][nt] = wmma32(af0, bf, acc[0][nt]);
      acc[1][nt] = wmma32(af1, bf, acc[1][nt]);
    }
  }

  // Epilogue: bias (+ RoPE), store. C/D layout: element r lives at
  // row = wave*32 + rt*16 + r + 8*hh, col = nt*16 + m16 within the block tile.
#pragma unroll
  for (int rt = 0; rt < 2; ++rt) {
    float vals[4][8];
#pragma unroll
    for (int nt = 0; nt < 4; ++nt) {
      float bb = bias[nBase + nt * 16 + m16];
#pragma unroll
      for (int r = 0; r < 8; ++r) vals[nt][r] = acc[rt][nt][r] + bb;
    }
    if (ROPE) {
      // Wave strip covers exactly one 64-wide head: rotate-half partner
      // (d ^ 32) sits in accumulator nt^2, same lane, same row.
      float rv[4][8];
#pragma unroll
      for (int nt = 0; nt < 4; ++nt) {
#pragma unroll
        for (int r = 0; r < 8; ++r) {
          int row = mBase + wave * 32 + rt * 16 + r + 8 * hh;
          int t   = row % Tdim;
          int col = nBase + nt * 16 + m16;
          int d   = col & 63;
          float cs = cosE[t * 64 + d];
          float sn = sinE[t * 64 + d];
          float part = vals[nt ^ 2][r];
          rv[nt][r] = vals[nt][r] * cs + ((d < 32) ? -part : part) * sn;
        }
      }
#pragma unroll
      for (int nt = 0; nt < 4; ++nt)
#pragma unroll
        for (int r = 0; r < 8; ++r) vals[nt][r] = rv[nt][r];
    }
#pragma unroll
    for (int nt = 0; nt < 4; ++nt) {
#pragma unroll
      for (int r = 0; r < 8; ++r) {
        int row = mBase + wave * 32 + rt * 16 + r + 8 * hh;
        size_t idx = (size_t)row * N + nBase + nt * 16 + m16;
        if (F16OUT) ((_Float16*)outP)[idx] = (_Float16)vals[nt][r];
        else        ((float*)outP)[idx]    = vals[nt][r];
      }
    }
  }
}

// ---------------------------------------------------------------------------
// Flash attention (causal, GQA): one block per (b, h, 64 q-rows); 4 waves,
// each owns 16 q-rows. 64-key j-blocks (16 WMMAs per online-softmax pass).
// ---------------------------------------------------------------------------
__global__ __launch_bounds__(128) void gqa_attn(
    const _Float16* __restrict__ Q, const _Float16* __restrict__ K,
    const _Float16* __restrict__ V, float* __restrict__ attnOut) {
  __shared__ alignas(16) _Float16 Kt[64][64];      // [j][d]  (= B-layout for QK^T)
  __shared__ alignas(16) _Float16 Vt[64][64];      // [d][j]  (= B-layout for P*V)
  __shared__ alignas(16) _Float16 Pb[4][16][64];   // per-wave P round-trip

  const int b  = blockIdx.z;
  const int h  = blockIdx.y;
  const int qb = blockIdx.x;
  const int kv = h >> 2;  // G = 4
  const int tid  = threadIdx.x;
  const int wave = tid >> 5;
  const int lane = tid & 31;
  const int m16  = lane & 15;
  const int hh   = lane >> 4;
  const int qbase = qb * 64;
  const int qw    = qbase + wave * 16;

  // Preload this wave's Q rows as two A-fragments (d = 0..31, 32..63).
  const _Float16* qrow = Q + (size_t)(b * T_ + qw + m16) * E_ + h * D_;
  v16h aq0 = load_afrag(qrow, hh);
  v16h aq1 = load_afrag(qrow + 32, hh);

  v8f accO[4];
  float rowMax[8], rowSum[8];
#pragma unroll
  for (int dt = 0; dt < 4; ++dt)
#pragma unroll
    for (int r = 0; r < 8; ++r) accO[dt][r] = 0.0f;
#pragma unroll
  for (int r = 0; r < 8; ++r) { rowMax[r] = -1e30f; rowSum[r] = 0.0f; }

  const int jblocks = qb + 1;  // j <= qbase+63 (block-uniform bound)
  for (int jb = 0; jb < jblocks; ++jb) {
    const int jbase = jb * 64;
    __syncthreads();
    // Stage K tile via async global->LDS: 64 rows x 64 halfs = 512 x 16B.
#pragma unroll
    for (int i = 0; i < 4; ++i) {
      int u = tid + i * 128;
      int row = u >> 3, seg = u & 7;
      async_copy16(&Kt[row][seg * 8],
                   K + (size_t)(b * T_ + jbase + row) * KVE_ + kv * D_ + seg * 8);
    }
    // Stage V tile transposed: Vt[d][j] (u32 global reads, 2 halfs per read).
#pragma unroll
    for (int i = 0; i < 16; ++i) {
      int idx = tid + i * 128;       // 2048 u32 = 64 rows x 32 pairs
      int j = idx >> 5, dp = (idx & 31) * 2;
      union { unsigned u; _Float16 hx[2]; } c;
      c.u = *(const unsigned*)(V + (size_t)(b * T_ + jbase + j) * KVE_ + kv * D_ + dp);
      Vt[dp][j]     = c.hx[0];
      Vt[dp + 1][j] = c.hx[1];
    }
    async_wait0();
    __syncthreads();

    // S = Q*K^T / sqrt(D) for four 16x16 j-tiles.
    v8f s[4];
#pragma unroll
    for (int jt = 0; jt < 4; ++jt) {
      v8f a;
#pragma unroll
      for (int r = 0; r < 8; ++r) a[r] = 0.0f;
      a = wmma32(aq0, load_bfrag(&Kt[jt * 16 + m16][0], hh), a);
      a = wmma32(aq1, load_bfrag(&Kt[jt * 16 + m16][32], hh), a);
      s[jt] = a;
    }
    // Scale + causal mask.
#pragma unroll
    for (int jt = 0; jt < 4; ++jt) {
#pragma unroll
      for (int r = 0; r < 8; ++r) {
        int qg = qw + r + 8 * hh;
        int jg = jbase + jt * 16 + m16;
        float v = s[jt][r] * 0.125f;  // 1/sqrt(64)
        s[jt][r] = (jg > qg) ? -1e30f : v;
      }
    }
    // Online softmax per row (row lives on the 16 lanes of a half-wave).
#pragma unroll
    for (int r = 0; r < 8; ++r) {
      float mx = fmaxf(fmaxf(s[0][r], s[1][r]), fmaxf(s[2][r], s[3][r]));
#pragma unroll
      for (int o = 1; o < 16; o <<= 1) mx = fmaxf(mx, __shfl_xor(mx, o, 32));
      float nm = fmaxf(rowMax[r], mx);
      float alpha = __expf(rowMax[r] - nm);
      rowMax[r] = nm;
      float p[4], sm = 0.0f;
#pragma unroll
      for (int jt = 0; jt < 4; ++jt) { p[jt] = __expf(s[jt][r] - nm); sm += p[jt]; }
#pragma unroll
      for (int o = 1; o < 16; o <<= 1) sm += __shfl_xor(sm, o, 32);
      rowSum[r] = rowSum[r] * alpha + sm;
#pragma unroll
      for (int dt = 0; dt < 4; ++dt) accO[dt][r] *= alpha;
      // Stash P in C-layout; re-read below in A-layout.
#pragma unroll
      for (int jt = 0; jt < 4; ++jt)
        Pb[wave][r + 8 * hh][jt * 16 + m16] = (_Float16)p[jt];
    }
    __syncthreads();

    // O += P * V  (P as two 16x32 A-frags, V tiles as 32x16 B-frags).
    v16h pf0 = load_afrag(&Pb[wave][m16][0], hh);
    v16h pf1 = load_afrag(&Pb[wave][m16][32], hh);
#pragma unroll
    for (int dt = 0; dt < 4; ++dt) {
      accO[dt] = wmma32(pf0, load_bfrag(&Vt[dt * 16 + m16][0], hh), accO[dt]);
      accO[dt] = wmma32(pf1, load_bfrag(&Vt[dt * 16 + m16][32], hh), accO[dt]);
    }
  }

  // Normalize + store attn (f32) in [b*T+q][h*64+d] layout.
#pragma unroll
  for (int dt = 0; dt < 4; ++dt) {
#pragma unroll
    for (int r = 0; r < 8; ++r) {
      int qg = qw + r + 8 * hh;
      float v = accO[dt][r] / rowSum[r];
      attnOut[(size_t)(b * T_ + qg) * E_ + h * D_ + dt * 16 + m16] = v;
    }
  }
}

// ---------------------------------------------------------------------------
// Host-side orchestration.
// ---------------------------------------------------------------------------
extern "C" void kernel_launch(void* const* d_in, const int* in_sizes, int n_in,
                              void* d_out, int out_size, void* d_ws, size_t ws_size,
                              hipStream_t stream) {
  (void)in_sizes; (void)n_in; (void)out_size; (void)ws_size;
  const float* query = (const float*)d_in[0];
  const float* key   = (const float*)d_in[1];
  const float* value = (const float*)d_in[2];
  const float* cosE  = (const float*)d_in[3];
  const float* sinE  = (const float*)d_in[4];
  // d_in[5] = mask: causal mask computed analytically, not read.
  const float* Wq = (const float*)d_in[6];
  const float* bq = (const float*)d_in[7];
  const float* Wk = (const float*)d_in[8];
  const float* bk = (const float*)d_in[9];
  const float* Wv = (const float*)d_in[10];
  const float* bv = (const float*)d_in[11];
  const float* Wo = (const float*)d_in[12];
  const float* bo = (const float*)d_in[13];

  // Workspace carve-up (256B aligned): 33 MiB total.
  size_t off = 0;
  auto carve = [&](size_t bytes) -> void* {
    void* p = (char*)d_ws + off;
    off += (bytes + 255) & ~(size_t)255;
    return p;
  };
  _Float16* WqT = (_Float16*)carve((size_t)E_ * E_ * 2);
  _Float16* WkT = (_Float16*)carve((size_t)E_ * KVE_ * 2);
  _Float16* WvT = (_Float16*)carve((size_t)E_ * KVE_ * 2);
  _Float16* WoT = (_Float16*)carve((size_t)E_ * E_ * 2);
  _Float16* Qh  = (_Float16*)carve((size_t)BT_ * E_ * 2);
  _Float16* Kh  = (_Float16*)carve((size_t)BT_ * KVE_ * 2);
  _Float16* Vh  = (_Float16*)carve((size_t)BT_ * KVE_ * 2);
  float*    attn = (float*)carve((size_t)BT_ * E_ * 4);

  // 1) Weights -> f16 transposed ([n][k]) for contiguous B-fragment K-pairs.
  transpose_w<<<(E_ * E_ + 255) / 256, 256, 0, stream>>>(Wq, WqT, E_, E_);
  transpose_w<<<(E_ * KVE_ + 255) / 256, 256, 0, stream>>>(Wk, WkT, E_, KVE_);
  transpose_w<<<(E_ * KVE_ + 255) / 256, 256, 0, stream>>>(Wv, WvT, E_, KVE_);
  transpose_w<<<(E_ * E_ + 255) / 256, 256, 0, stream>>>(Wo, WoT, E_, E_);

  // 2) QKV projections (fused bias; RoPE for Q and K), outputs f16.
  gqa_gemm<true, true><<<dim3(E_ / 64, BT_ / 128), 128, 0, stream>>>(
      query, WqT, bq, Qh, BT_, E_, E_, T_, cosE, sinE);
  gqa_gemm<true, true><<<dim3(KVE_ / 64, BT_ / 128), 128, 0, stream>>>(
      key, WkT, bk, Kh, BT_, KVE_, E_, T_, cosE, sinE);
  gqa_gemm<false, true><<<dim3(KVE_ / 64, BT_ / 128), 128, 0, stream>>>(
      value, WvT, bv, Vh, BT_, KVE_, E_, T_, nullptr, nullptr);

  // 3) Causal GQA flash attention -> attn (f32, [B*T][E]).
  gqa_attn<<<dim3(T_ / 64, H_, B_), 128, 0, stream>>>(Qh, Kh, Vh, attn);

  // 4) Output projection -> d_out (f32) + bo.
  gqa_gemm<false, false><<<dim3(E_ / 64, BT_ / 128), 128, 0, stream>>>(
      attn, WoT, bo, d_out, BT_, E_, E_, T_, nullptr, nullptr);
}
